// LLM_18923625906329
// MI455X (gfx1250) — matrix operand order repaired and checked by reading, not compiled, not run on hardware
//
#include <hip/hip_runtime.h>

// ---------------------------------------------------------------------------
// Model constants (match reference)
// ---------------------------------------------------------------------------
constexpr int kB   = 2;
constexpr int kT   = 2048;
constexpr int kD   = 1024;
constexpr int kH   = 16;
constexpr int kHD  = 64;
constexpr int kL   = 8;
constexpr int kHID = 2752;
constexpr int kV   = 32000;
constexpr int kM   = kB * kT;       // 4096 tokens
static_assert(kM % 128 == 0, "M tiles exact");

typedef __attribute__((ext_vector_type(16))) __bf16 bf16x16;
typedef __attribute__((ext_vector_type(8)))  __bf16 bf16x8;
typedef __attribute__((ext_vector_type(8)))  float  f32x8;
typedef int v4i __attribute__((vector_size(16)));

#ifndef __has_builtin
#define __has_builtin(x) 0
#endif
#if __has_builtin(__builtin_amdgcn_global_load_async_to_lds_b128) && \
    __has_builtin(__builtin_amdgcn_s_wait_asynccnt)
#define USE_ASYNC_COPY 1
#else
#define USE_ASYNC_COPY 0
#endif

#define AS1 __attribute__((address_space(1)))
#define AS3 __attribute__((address_space(3)))

// ---------------------------------------------------------------------------
// Tiled NT GEMM:  C[M,N] = A_bf16[M,K] (row-major) x W_f32[N,K]^T  (fp32 acc)
// mode 0: C f32   | mode 1: C f32 = acc + R (residual) | mode 2: C bf16
// Block tile 128x128x32, 256 threads = 8 waves (2x4), wave tile 64x32.
// Double-buffered LDS; A tile staged with async global->LDS DMA when available.
// Out-of-range N columns are CLAMPED (not branched): garbage only lands in
// accumulator columns that are masked at the final store.
// ---------------------------------------------------------------------------
__global__ __launch_bounds__(256)
void gemm_nt_kernel(const __bf16* __restrict__ A, const float* __restrict__ W,
                    void* __restrict__ Cout, const float* __restrict__ R,
                    int M, int N, int K, int mode)
{
    constexpr int BM = 128, BN = 128, BK = 32, PA = 8, PB = 8;
    __shared__ __bf16 As[2][BM][BK + PA];   // row-major A tile (80B stride)
    __shared__ __bf16 Bs[2][BK][BN + PB];   // k-major  W tile (272B stride)

    const int tid  = threadIdx.x;
    const int m0   = blockIdx.y * BM;
    const int n0   = blockIdx.x * BN;
    const int wave = tid >> 5, lane = tid & 31;
    const int wm   = wave >> 2, wn = wave & 3;      // 2 x 4 wave grid
    const int l16  = lane & 15, hf = lane >> 4;
    const int KT   = K / BK;

    // per-thread staging coordinates (constant across k-steps)
    const int arow0 = (tid) >> 2,        ac0 = ((tid) & 3) * 8;
    const int arow1 = (tid + 256) >> 2,  ac1 = ((tid + 256) & 3) * 8;

    auto stageA = [&](int kt, int buf) {
        const __bf16* g0 = A + (size_t)(m0 + arow0) * K + (size_t)kt * BK + ac0;
        const __bf16* g1 = A + (size_t)(m0 + arow1) * K + (size_t)kt * BK + ac1;
#if USE_ASYNC_COPY
        __builtin_amdgcn_global_load_async_to_lds_b128(
            (AS1 v4i*)(AS1 void*)(uintptr_t)g0,
            (AS3 v4i*)(AS3 void*)&As[buf][arow0][ac0], 0, 0);
        __builtin_amdgcn_global_load_async_to_lds_b128(
            (AS1 v4i*)(AS1 void*)(uintptr_t)g1,
            (AS3 v4i*)(AS3 void*)&As[buf][arow1][ac1], 0, 0);
#else
        *(bf16x8*)&As[buf][arow0][ac0] = *(const bf16x8*)g0;
        *(bf16x8*)&As[buf][arow1][ac1] = *(const bf16x8*)g1;
#endif
    };
    auto stageB = [&](int kt, int buf) {
#pragma unroll
        for (int i = 0; i < 4; ++i) {
            int c  = tid + i * 256;
            int n  = c >> 3, k4 = (c & 7) * 4;
            int gn = n0 + n;
            if (gn >= N) gn = N - 1;               // clamp: column masked at store
            const float* wp = W + (size_t)gn * K + (size_t)kt * BK + k4;
            if (i == 0 && kt + 2 < KT) __builtin_prefetch(wp + 2 * BK, 0, 1);
            float4 w = *(const float4*)wp;
            Bs[buf][k4 + 0][n] = (__bf16)w.x;
            Bs[buf][k4 + 1][n] = (__bf16)w.y;
            Bs[buf][k4 + 2][n] = (__bf16)w.z;
            Bs[buf][k4 + 3][n] = (__bf16)w.w;
        }
    };

    f32x8 acc[4][2];
#pragma unroll
    for (int i = 0; i < 4; ++i)
#pragma unroll
        for (int j = 0; j < 2; ++j) acc[i][j] = {};

    stageA(0, 0);
    stageB(0, 0);

    int cur = 0;
    for (int kt = 0; kt < KT; ++kt) {
#if USE_ASYNC_COPY
        __builtin_amdgcn_s_wait_asynccnt(0);       // DMA into buf `cur` done
#endif
        __syncthreads();                            // tile `cur` visible to all
        if (kt + 1 < KT) {                          // overlap next stage w/ WMMA
            stageA(kt + 1, cur ^ 1);
            stageB(kt + 1, cur ^ 1);
        }

        // ---- A fragments: lane<->M row, element<->K (interleaved halves) ----
        bf16x16 af[4];
#pragma unroll
        for (int mf = 0; mf < 4; ++mf) {
            const __bf16* p = &As[cur][wm * 64 + mf * 16 + l16][hf * 8];
            bf16x8 lo = *(const bf16x8*)p;
            bf16x8 hi = *(const bf16x8*)(p + 16);
#pragma unroll
            for (int j = 0; j < 8; ++j) { af[mf][j] = lo[j]; af[mf][8 + j] = hi[j]; }
        }
        // ---- B fragments: lane<->N, element<->K (interleaved halves) ----
        bf16x16 bfr[2];
#pragma unroll
        for (int nf = 0; nf < 2; ++nf) {
            const __bf16* p = &Bs[cur][hf * 8][wn * 32 + nf * 16 + l16];
#pragma unroll
            for (int j = 0; j < 8; ++j) { bfr[nf][j] = p[j * (BN + PB)]; bfr[nf][8 + j] = p[(16 + j) * (BN + PB)]; }
        }
#pragma unroll
        for (int mf = 0; mf < 4; ++mf)
#pragma unroll
            for (int nf = 0; nf < 2; ++nf)
                acc[mf][nf] = __builtin_amdgcn_wmma_f32_16x16x32_bf16(
                    false, af[mf], false, bfr[nf], (short)0, acc[mf][nf], false, false);
        cur ^= 1;
    }

    // ---- epilogue: C layout lane<->N, reg r (+8*half) <-> M ----
#pragma unroll
    for (int mf = 0; mf < 4; ++mf) {
        int mb = m0 + wm * 64 + mf * 16 + hf * 8;
#pragma unroll
        for (int nf = 0; nf < 2; ++nf) {
            int gn = n0 + wn * 32 + nf * 16 + l16;
            if (gn < N) {
#pragma unroll
                for (int r = 0; r < 8; ++r) {
                    float  v   = acc[mf][nf][r];
                    size_t off = (size_t)(mb + r) * N + gn;
                    if (mode == 0)      ((float*)Cout)[off]  = v;
                    else if (mode == 1) ((float*)Cout)[off]  = v + R[off];
                    else                ((__bf16*)Cout)[off] = (__bf16)v;
                }
            }
        }
    }
    (void)M;
}

// ---------------------------------------------------------------------------
// Flash attention, causal.  Q/K/V bf16 [B*H, T, HD], O bf16 [M, D].
// Grid (T/64, B*H), 128 threads = 4 waves; each wave owns 16 q-rows.
// ---------------------------------------------------------------------------
__global__ __launch_bounds__(128)
void flash_attn_kernel(const __bf16* __restrict__ Q, const __bf16* __restrict__ Km,
                       const __bf16* __restrict__ Vm, __bf16* __restrict__ O)
{
    constexpr int PAD = 8;
    __shared__ __bf16 Ks[kHD][64 + PAD];   // K tile transposed: [hd][kv]
    __shared__ __bf16 Ps[64][64 + PAD];    // P tile: [q-row][kv]

    const int tid  = threadIdx.x;
    const int wave = tid >> 5, lane = tid & 31;
    const int l16  = lane & 15, hf = lane >> 4;
    const int bh   = blockIdx.y;
    const int qt   = blockIdx.x;
    const int q0   = qt * 64;
    const size_t base = (size_t)bh * kT * kHD;
    const float scale = 0.125f;            // HD^-0.5

    // Q fragments for this wave's 16 rows (kept in regs for the whole kv loop)
    bf16x16 qa[2];
    const int qrow = q0 + wave * 16 + l16;
#pragma unroll
    for (int fk = 0; fk < 2; ++fk) {
        const __bf16* p = Q + base + (size_t)qrow * kHD + fk * 32 + hf * 8;
        bf16x8 lo = *(const bf16x8*)p;
        bf16x8 hi = *(const bf16x8*)(p + 16);
#pragma unroll
        for (int j = 0; j < 8; ++j) { qa[fk][j] = lo[j]; qa[fk][8 + j] = hi[j]; }
    }

    float mrow[8], lrow[8];
    f32x8 oacc[4];
#pragma unroll
    for (int r = 0; r < 8; ++r) { mrow[r] = -1e30f; lrow[r] = 0.f; }
#pragma unroll
    for (int nf = 0; nf < 4; ++nf) oacc[nf] = {};

    const int qbase = q0 + wave * 16 + hf * 8;   // + r gives this lane's q index

    for (int kt = 0; kt <= qt; ++kt) {
        // ---- stage K tile transposed into LDS ----
#pragma unroll
        for (int i = 0; i < 4; ++i) {
            int c  = tid + i * 128;
            int kr = c >> 3, c8 = (c & 7) * 8;
            bf16x8 v = *(const bf16x8*)(Km + base + (size_t)(kt * 64 + kr) * kHD + c8);
#pragma unroll
            for (int j = 0; j < 8; ++j) Ks[c8 + j][kr] = v[j];
        }
        __syncthreads();

        // ---- S = Q K^T : per wave 16x64, 8 WMMAs ----
        f32x8 sacc[4];
#pragma unroll
        for (int nf = 0; nf < 4; ++nf) sacc[nf] = {};
#pragma unroll
        for (int fk = 0; fk < 2; ++fk) {
#pragma unroll
            for (int nf = 0; nf < 4; ++nf) {
                const __bf16* p = &Ks[fk * 32 + hf * 8][nf * 16 + l16];
                bf16x16 kb;
#pragma unroll
                for (int j = 0; j < 8; ++j) { kb[j] = p[j * (64 + PAD)]; kb[8 + j] = p[(16 + j) * (64 + PAD)]; }
                sacc[nf] = __builtin_amdgcn_wmma_f32_16x16x32_bf16(
                    false, qa[fk], false, kb, (short)0, sacc[nf], false, false);
            }
        }

        // ---- online softmax (fp32, in C-register layout) ----
#pragma unroll
        for (int r = 0; r < 8; ++r) {
            float mx = -1e30f;
#pragma unroll
            for (int nf = 0; nf < 4; ++nf) {
                float s  = sacc[nf][r] * scale;
                int   kv = kt * 64 + nf * 16 + l16;
                if (kv > qbase + r) s = -1e30f;      // causal mask
                sacc[nf][r] = s;
                mx = fmaxf(mx, s);
            }
#pragma unroll
            for (int off = 8; off >= 1; off >>= 1) mx = fmaxf(mx, __shfl_xor(mx, off, 16));
            float mnew  = fmaxf(mrow[r], mx);
            float alpha = __expf(mrow[r] - mnew);
            float psum  = 0.f;
#pragma unroll
            for (int nf = 0; nf < 4; ++nf) {
                float pv = __expf(sacc[nf][r] - mnew);
                sacc[nf][r] = pv;
                psum += pv;
            }
#pragma unroll
            for (int off = 8; off >= 1; off >>= 1) psum += __shfl_xor(psum, off, 16);
            lrow[r] = lrow[r] * alpha + psum;
            mrow[r] = mnew;
#pragma unroll
            for (int nf = 0; nf < 4; ++nf) oacc[nf][r] = oacc[nf][r] * alpha;
            // P (bf16) to LDS so it can be reloaded in A-matrix layout
            int prow = wave * 16 + hf * 8 + r;
#pragma unroll
            for (int nf = 0; nf < 4; ++nf)
                Ps[prow][nf * 16 + l16] = (__bf16)sacc[nf][r];
        }

        // ---- O += P V : P from LDS (A layout), V direct from global (B layout) ----
#pragma unroll
        for (int fk = 0; fk < 2; ++fk) {
            bf16x16 pa;
            const __bf16* pp = &Ps[wave * 16 + l16][fk * 32 + hf * 8];
            bf16x8 lo = *(const bf16x8*)pp;
            bf16x8 hi = *(const bf16x8*)(pp + 16);
#pragma unroll
            for (int j = 0; j < 8; ++j) { pa[j] = lo[j]; pa[8 + j] = hi[j]; }
#pragma unroll
            for (int nf = 0; nf < 4; ++nf) {
                const __bf16* vp = Vm + base + (size_t)(kt * 64 + fk * 32 + hf * 8) * kHD + nf * 16 + l16;
                bf16x16 vb;
#pragma unroll
                for (int j = 0; j < 8; ++j) { vb[j] = vp[j * kHD]; vb[8 + j] = vp[(16 + j) * kHD]; }
                oacc[nf] = __builtin_amdgcn_wmma_f32_16x16x32_bf16(
                    false, pa, false, vb, (short)0, oacc[nf], false, false);
            }
        }
        __syncthreads();
    }

    // ---- write O (normalized) into [M, D] bf16 layout ----
    const int b = bh / kH, h = bh % kH;
#pragma unroll
    for (int r = 0; r < 8; ++r) {
        int    t   = qbase + r;
        size_t m   = (size_t)b * kT + t;
        float  inv = 1.f / lrow[r];
#pragma unroll
        for (int nf = 0; nf < 4; ++nf)
            O[m * kD + h * kHD + nf * 16 + l16] = (__bf16)(oacc[nf][r] * inv);
    }
}

// ---------------------------------------------------------------------------
// Elementwise kernels
// ---------------------------------------------------------------------------
__global__ __launch_bounds__(256)
void embed_kernel(const int* __restrict__ ids, const float* __restrict__ emb,
                  float* __restrict__ X)
{
    int m = blockIdx.x;
    const float* src = emb + (size_t)ids[m] * kD;
    float* dst = X + (size_t)m * kD;
    for (int i = threadIdx.x; i < kD; i += 256) dst[i] = src[i];
}

__global__ __launch_bounds__(256)
void rmsnorm_kernel(const float* __restrict__ X, const float* __restrict__ W,
                    __bf16* __restrict__ Y)
{
    __shared__ float wsum[8];
    int m = blockIdx.x;
    const float* row = X + (size_t)m * kD;
    float ss = 0.f;
    for (int i = threadIdx.x; i < kD; i += 256) { float v = row[i]; ss += v * v; }
#pragma unroll
    for (int off = 16; off >= 1; off >>= 1) ss += __shfl_xor(ss, off, 32);
    if ((threadIdx.x & 31) == 0) wsum[threadIdx.x >> 5] = ss;
    __syncthreads();
    float tot = 0.f;
#pragma unroll
    for (int i = 0; i < 8; ++i) tot += wsum[i];
    float rinv = rsqrtf(tot * (1.f / kD) + 1e-6f);
    for (int i = threadIdx.x; i < kD; i += 256)
        Y[(size_t)m * kD + i] = (__bf16)(row[i] * rinv * W[i]);
}

__global__ __launch_bounds__(256)
void rope_split_kernel(const float* __restrict__ qkv, __bf16* __restrict__ Qo,
                       __bf16* __restrict__ Ko, __bf16* __restrict__ Vo)
{
    int m = blockIdx.x;
    int b = m / kT, t = m % kT;
    const float* row = qkv + (size_t)m * 3 * kD;
    for (int i = threadIdx.x; i < kD; i += 256) {
        int h = i >> 6, hd = i & 63, j = hd & 31;
        float inv = __powf(10000.f, -(float)(2 * j) * (1.f / 64.f));
        float ang = (float)t * inv;
        float c = __cosf(ang), s = __sinf(ang);
        int   ip  = (hd < 32) ? i + 32 : i - 32;
        float sgn = (hd < 32) ? -1.f : 1.f;
        float qv = row[i],        kv = row[kD + i],  vv = row[2 * kD + i];
        float qp = row[ip],       kp = row[kD + ip];
        float qr = qv * c + sgn * qp * s;
        float kr = kv * c + sgn * kp * s;
        size_t dst = ((size_t)(b * kH + h) * kT + t) * kHD + hd;
        Qo[dst] = (__bf16)qr; Ko[dst] = (__bf16)kr; Vo[dst] = (__bf16)vv;
    }
}

__global__ __launch_bounds__(256)
void silu_mul_kernel(const __bf16* __restrict__ g, const __bf16* __restrict__ v,
                     __bf16* __restrict__ o, int n)
{
    int i = blockIdx.x * 256 + threadIdx.x;
    if (i < n) {
        float gv = (float)g[i], vv = (float)v[i];
        o[i] = (__bf16)(gv * vv / (1.f + __expf(-gv)));
    }
}

// ---------------------------------------------------------------------------
// Orchestration
// ---------------------------------------------------------------------------
extern "C" void kernel_launch(void* const* d_in, const int* in_sizes, int n_in,
                              void* d_out, int out_size, void* d_ws, size_t ws_size,
                              hipStream_t stream)
{
    (void)in_sizes; (void)n_in; (void)out_size; (void)ws_size;
    const int*   ids  = (const int*)  d_in[0];
    const float* emb  = (const float*)d_in[1];
    const float* n1w  = (const float*)d_in[2];
    const float* qkvw = (const float*)d_in[3];
    const float* outw = (const float*)d_in[4];
    const float* n2w  = (const float*)d_in[5];
    const float* w1w  = (const float*)d_in[6];
    const float* w2w  = (const float*)d_in[7];
    const float* w3w  = (const float*)d_in[8];
    const float* now  = (const float*)d_in[9];

    char*  ws  = (char*)d_ws;
    size_t off = 0;
    auto alloc = [&](size_t bytes) -> void* {
        void* p = ws + off;
        off += (bytes + 255) & ~(size_t)255;
        return p;
    };
    float*  x    = (float*) alloc((size_t)kM * kD * 4);
    __bf16* xn   = (__bf16*)alloc((size_t)kM * kD * 2);
    float*  qkv  = (float*) alloc((size_t)kM * 3 * kD * 4);
    __bf16* qb   = (__bf16*)alloc((size_t)kM * kD * 2);
    __bf16* kbuf = (__bf16*)alloc((size_t)kM * kD * 2);
    __bf16* vbuf = (__bf16*)alloc((size_t)kM * kD * 2);
    __bf16* ob   = (__bf16*)alloc((size_t)kM * kD * 2);
    __bf16* gb   = (__bf16*)alloc((size_t)kM * kHID * 2);
    __bf16* ub   = (__bf16*)alloc((size_t)kM * kHID * 2);
    __bf16* hb   = (__bf16*)alloc((size_t)kM * kHID * 2);

    const dim3 gQKV((3 * kD + 127) / 128, kM / 128);
    const dim3 gOUT((kD + 127) / 128, kM / 128);
    const dim3 gFFN((kHID + 127) / 128, kM / 128);
    const dim3 gLM ((kV + 127) / 128, kM / 128);
    const dim3 gATT(kT / 64, kB * kH);

    embed_kernel<<<kM, 256, 0, stream>>>(ids, emb, x);
    for (int l = 0; l < kL; ++l) {
        rmsnorm_kernel<<<kM, 256, 0, stream>>>(x, n1w + (size_t)l * kD, xn);
        gemm_nt_kernel<<<gQKV, 256, 0, stream>>>(xn, qkvw + (size_t)l * 3 * kD * kD,
                                                 qkv, nullptr, kM, 3 * kD, kD, 0);
        rope_split_kernel<<<kM, 256, 0, stream>>>(qkv, qb, kbuf, vbuf);
        flash_attn_kernel<<<gATT, 128, 0, stream>>>(qb, kbuf, vbuf, ob);
        gemm_nt_kernel<<<gOUT, 256, 0, stream>>>(ob, outw + (size_t)l * kD * kD,
                                                 x, x, kM, kD, kD, 1);
        rmsnorm_kernel<<<kM, 256, 0, stream>>>(x, n2w + (size_t)l * kD, xn);
        gemm_nt_kernel<<<gFFN, 256, 0, stream>>>(xn, w1w + (size_t)l * kHID * kD,
                                                 gb, nullptr, kM, kHID, kD, 2);
        gemm_nt_kernel<<<gFFN, 256, 0, stream>>>(xn, w2w + (size_t)l * kHID * kD,
                                                 ub, nullptr, kM, kHID, kD, 2);
        silu_mul_kernel<<<((size_t)kM * kHID + 255) / 256, 256, 0, stream>>>(
            gb, ub, hb, kM * kHID);
        gemm_nt_kernel<<<gOUT, 256, 0, stream>>>(hb, w3w + (size_t)l * kD * kHID,
                                                 x, x, kM, kD, kHID, 1);
    }
    rmsnorm_kernel<<<kM, 256, 0, stream>>>(x, now, xn);
    gemm_nt_kernel<<<gLM, 256, 0, stream>>>(xn, emb, d_out, nullptr, kM, kV, kD, 0);
}
